// SelfAttention_21552145891525
// MI455X (gfx1250) — matrix operand order repaired and checked by reading, not compiled, not run on hardware
//
#include <hip/hip_runtime.h>
#include <hip/hip_bf16.h>

typedef __attribute__((ext_vector_type(16))) _Float16     v16h;
typedef __attribute__((ext_vector_type(8)))  _Float16     v8h;
typedef __attribute__((ext_vector_type(8)))  float        v8f;
typedef __attribute__((ext_vector_type(4)))  unsigned int v4u;
typedef __attribute__((ext_vector_type(8)))  int          v8i;
typedef __attribute__((ext_vector_type(4)))  int          v4i;

union F16x16 {
    v16h v;
    v8h  h2[2];
    _Float16 h[16];
};

#define WMMA_F16(a, b, c) \
    __builtin_amdgcn_wmma_f32_16x16x32_f16(false, (a), false, (b), (short)0, (c), false, false)

// ---------------------------------------------------------------------------
// TDM 2-D tile load: build D# group0/group1 per CDNA5 ISA 8.3/8.4 and issue
// tensor_load_to_lds (6-arg form; groups 2/3 zero -> 2-D tensor, tile_dim2..4
// unused).  data_size = 2 bytes (f16).
// ---------------------------------------------------------------------------
__device__ __forceinline__ void tdm_load_2d_f16(unsigned int lds_addr,
                                                const void* gptr,
                                                unsigned int tensor_d0,
                                                unsigned int tensor_d1,
                                                unsigned int tile_d0,
                                                unsigned int tile_d1,
                                                unsigned int stride_d0) {
    unsigned long long ga = (unsigned long long)(size_t)gptr;
    v4u g0;
    g0[0] = 1u;                                        // count=1, user descriptor
    g0[1] = lds_addr;                                  // LDS byte address
    g0[2] = (unsigned int)(ga & 0xffffffffu);          // global_addr[31:0]
    g0[3] = (unsigned int)((ga >> 32) & 0x01ffffffu)   // global_addr[56:32]
          | 0x80000000u;                               // type=2 ("image") in [127:126]
    v8i g1;
    g1[0] = (int)(1u << 16);                           // workgroup_mask=0, data_size=1 (2B)
    g1[1] = (int)((tensor_d0 & 0xffffu) << 16);        // tensor_dim0[15:0] at bit48
    g1[2] = (int)((tensor_d0 >> 16) | ((tensor_d1 & 0xffffu) << 16));
    g1[3] = (int)((tensor_d1 >> 16) | (tile_d0 << 16)); // tile_dim0 at bit112
    g1[4] = (int)(tile_d1 & 0xffffu);                  // tile_dim1; tile_dim2=0 unused
    g1[5] = (int)stride_d0;                            // tensor_dim0_stride[31:0]
    g1[6] = 0;
    g1[7] = 0;
    v4i z4 = {0, 0, 0, 0};
    v8i z8 = {0, 0, 0, 0, 0, 0, 0, 0};
    __builtin_amdgcn_tensor_load_to_lds(g0, g1, z4, z4, z8, 0);
}

// ---------------------------------------------------------------------------
// Kernel 1a: f32 -> f16 elementwise convert
// ---------------------------------------------------------------------------
__global__ void cvt_f32_to_f16(const float* __restrict__ src,
                               _Float16* __restrict__ dst, int n) {
    int i = blockIdx.x * blockDim.x + threadIdx.x;
    if (i < n) dst[i] = (_Float16)src[i];
}

// ---------------------------------------------------------------------------
// Kernel 1b: transpose K x N f32 -> N x K f16 (so WMMA B-fragments are
// contiguous per-lane 128-bit loads)
// ---------------------------------------------------------------------------
__global__ void transpose_to_f16(const float* __restrict__ src,
                                 _Float16* __restrict__ dst, int K, int N) {
    int i = blockIdx.x * blockDim.x + threadIdx.x;
    if (i < K * N) {
        int k = i / N, n = i % N;
        dst[n * K + k] = (_Float16)src[i];
    }
}

// ---------------------------------------------------------------------------
// Kernel 2: QKV projection GEMM.  One wave computes a 16x64 output slab:
// each A fragment is reused across 4 N-tiles (4 WMMAs per A load).
// xh: (8192 x 64) f16.  wT: (512 x 64) f16 (N x K).  K=64 -> 2 k-steps.
// MODE 0: Q -> (b,h,l,d), pre-scaled by 1/sqrt(64)
// MODE 1: K -> (b,h,l,d)
// MODE 2: V -> (b,h,d,l)  (transposed so PV B-fragments are contiguous)
// ---------------------------------------------------------------------------
template <int MODE>
__global__ __launch_bounds__(32)
void qkv_gemm(const _Float16* __restrict__ xh,
              const _Float16* __restrict__ wT,
              _Float16* __restrict__ dst) {
    const int L  = threadIdx.x;
    const int g  = L >> 4;
    const int ln = L & 15;
    const int m0 = blockIdx.x * 16;
    const int n0 = blockIdx.y * 64;

    v8f c[4] = {};
#pragma unroll
    for (int ks = 0; ks < 2; ++ks) {
        const int k0 = ks * 32;
        F16x16 a;
        const _Float16* ap = xh + (size_t)(m0 + ln) * 64 + k0 + g * 8;
        a.h2[0] = *(const v8h*)(ap);
        a.h2[1] = *(const v8h*)(ap + 16);
#pragma unroll
        for (int t = 0; t < 4; ++t) {
            F16x16 b;
            const _Float16* bp = wT + (size_t)(n0 + t * 16 + ln) * 64 + k0 + g * 8;
            b.h2[0] = *(const v8h*)(bp);
            b.h2[1] = *(const v8h*)(bp + 16);
            c[t] = WMMA_F16(a.v, b.v, c[t]);
        }
    }

    const float scale = (MODE == 0) ? 0.125f : 1.0f;
#pragma unroll
    for (int t = 0; t < 4; ++t) {
#pragma unroll
        for (int r = 0; r < 8; ++r) {
            const int m = m0 + r + 8 * g;       // D layout: M = r + 8*half
            const int n = n0 + t * 16 + ln;     // N = lane%16
            const int bb = m >> 12;
            const int l  = m & 4095;
            const int h  = n >> 6;
            const int d  = n & 63;
            const _Float16 val = (_Float16)(c[t][r] * scale);
            if (MODE == 2) {
                dst[(size_t)(((bb * 8 + h) * 64 + d)) * 4096 + l] = val;
            } else {
                dst[(size_t)(((bb * 8 + h) * 4096 + l)) * 64 + d] = val;
            }
        }
    }
}

// ---------------------------------------------------------------------------
// Kernel 3: flash attention, 64-key tiles staged via TDM.
// 4 waves/block share one (b,h); each wave owns a 16-query tile and loops over
// 4096 keys in 64-key tiles.  Wave 0 issues two tensor_load_to_lds per tile
// (K tile 64x64, V tile 64x64) into double-buffered LDS; all 4 waves read
// WMMA B-fragments from LDS.  Prefetch index wraps so the steady-state loop
// is branch-free: always issue, always s_wait_tensorcnt(2).  All row loops
// fully unrolled (no v_movrels dynamic VGPR indexing).
// ---------------------------------------------------------------------------
__global__ __launch_bounds__(128)
void flash_attn(const _Float16* __restrict__ Qh,
                const _Float16* __restrict__ Kh,
                const _Float16* __restrict__ Vt,
                _Float16* __restrict__ attn) {
    __shared__ _Float16 Kbuf[2][64 * 64];   // keys x dims  (8 KB each)
    __shared__ _Float16 Vbuf[2][64 * 64];   // dims x keys  (8 KB each)
    __shared__ _Float16 Pbuf[4][16 * 64];   // per-wave P reshape slab (2 KB)

    const int L    = threadIdx.x & 31;
    const int wave = threadIdx.x >> 5;
    const int g  = L >> 4;
    const int ln = L & 15;
    const int bh = blockIdx.y;              // b*8 + h
    const int qt = blockIdx.x * 4 + wave;
    const int q0 = qt * 16;
    const int b  = bh >> 3;
    const int h  = bh & 7;

    const _Float16* Qb = Qh + (size_t)bh * 4096 * 64;
    const _Float16* Kb = Kh + (size_t)bh * 4096 * 64;
    const _Float16* Vb = Vt + (size_t)bh * 64 * 4096;

    // Q A-fragments (dims 0..31 / 32..63), loaded once per wave
    F16x16 a0, a1;
    {
        const _Float16* p = Qb + (size_t)(q0 + ln) * 64;
        a0.h2[0] = *(const v8h*)(p + g * 8);
        a0.h2[1] = *(const v8h*)(p + g * 8 + 16);
        a1.h2[0] = *(const v8h*)(p + 32 + g * 8);
        a1.h2[1] = *(const v8h*)(p + 32 + g * 8 + 16);
    }

    float mrow[8], lrow[8];
    v8f acc[4] = {};
#pragma unroll
    for (int r = 0; r < 8; ++r) { mrow[r] = -1e30f; lrow[r] = 0.0f; }

    _Float16* Pl = Pbuf[wave];
    const bool issuer = (threadIdx.x < 32);   // wave 0 drives the TDM

    // Prefetch tile 0 into buffer 0
    if (issuer) {
        tdm_load_2d_f16((unsigned int)(size_t)&Kbuf[0][0], Kb,
                        64, 4096, 64, 64, 64);
        tdm_load_2d_f16((unsigned int)(size_t)&Vbuf[0][0], Vb,
                        4096, 64, 64, 64, 4096);
    }

    for (int kt = 0; kt < 64; ++kt) {
        const int cur = kt & 1;
        if (issuer) {
            // Wrapping prefetch: last iteration re-fetches tile 0 into the
            // idle buffer (harmless; block-local), keeping this branch-free.
            const int kb1 = ((kt + 1) & 63) * 64;
            tdm_load_2d_f16((unsigned int)(size_t)&Kbuf[cur ^ 1][0],
                            Kb + (size_t)kb1 * 64, 64, 4096, 64, 64, 64);
            tdm_load_2d_f16((unsigned int)(size_t)&Vbuf[cur ^ 1][0],
                            Vb + kb1, 4096, 64, 64, 64, 4096);
            __builtin_amdgcn_s_wait_tensorcnt(2);  // tile kt done, kt+1 in flight
        }
        __syncthreads();   // LDS tile kt visible to all waves

        const _Float16* Kl = Kbuf[cur];
        const _Float16* Vl = Vbuf[cur];

        // S = Q * K^T : four 16x16 f32 key sub-tiles x two k-steps (8 WMMAs).
        // Batch the 4 fragment loads per k-step ahead of the 4 WMMAs so one
        // dscnt wait covers all four.
        v8f s[4] = {};
#pragma unroll
        for (int ks = 0; ks < 2; ++ks) {
            F16x16 bk[4];
#pragma unroll
            for (int j = 0; j < 4; ++j) {
                const _Float16* p = Kl + (size_t)(j * 16 + ln) * 64 + ks * 32 + g * 8;
                bk[j].h2[0] = *(const v8h*)(p);
                bk[j].h2[1] = *(const v8h*)(p + 16);
            }
#pragma unroll
            for (int j = 0; j < 4; ++j)
                s[j] = WMMA_F16(ks == 0 ? a0.v : a1.v, bk[j].v, s[j]);
        }

        // Online softmax: row M = r + 8*g spans lanes ln=0..15 of this half.
        // Fully unrolled: all register accesses static.
#pragma unroll
        for (int r = 0; r < 8; ++r) {
            float t = fmaxf(fmaxf(s[0][r], s[1][r]), fmaxf(s[2][r], s[3][r]));
            t = fmaxf(t, __shfl_xor(t, 8, 32));
            t = fmaxf(t, __shfl_xor(t, 4, 32));
            t = fmaxf(t, __shfl_xor(t, 2, 32));
            t = fmaxf(t, __shfl_xor(t, 1, 32));
            const float mn   = fmaxf(mrow[r], t);
            const float corr = __expf(mrow[r] - mn);
            mrow[r] = mn;
            float su = 0.0f;
#pragma unroll
            for (int j = 0; j < 4; ++j) {
                const float e = __expf(s[j][r] - mn);
                Pl[(r + 8 * g) * 64 + j * 16 + ln] = (_Float16)e;
                su += e;
            }
            su += __shfl_xor(su, 8, 32);
            su += __shfl_xor(su, 4, 32);
            su += __shfl_xor(su, 2, 32);
            su += __shfl_xor(su, 1, 32);
            lrow[r] = lrow[r] * corr + su;
#pragma unroll
            for (int cch = 0; cch < 4; ++cch) acc[cch][r] *= corr;
        }

        // P (16x64) as two A-fragments (keys 0..31 / 32..63) from the slab.
        F16x16 ap0, ap1;
        ap0.h2[0] = *(const v8h*)(Pl + ln * 64 + g * 8);
        ap0.h2[1] = *(const v8h*)(Pl + ln * 64 + g * 8 + 16);
        ap1.h2[0] = *(const v8h*)(Pl + ln * 64 + 32 + g * 8);
        ap1.h2[1] = *(const v8h*)(Pl + ln * 64 + 32 + g * 8 + 16);

        // O += P * V : 4 dim-chunks x 2 key k-steps (8 WMMAs).
#pragma unroll
        for (int cch = 0; cch < 4; ++cch) {
            const _Float16* vp = Vl + (size_t)(cch * 16 + ln) * 64;
            F16x16 bv0, bv1;
            bv0.h2[0] = *(const v8h*)(vp + g * 8);
            bv0.h2[1] = *(const v8h*)(vp + g * 8 + 16);
            bv1.h2[0] = *(const v8h*)(vp + 32 + g * 8);
            bv1.h2[1] = *(const v8h*)(vp + 32 + g * 8 + 16);
            acc[cch] = WMMA_F16(ap0.v, bv0.v, acc[cch]);
            acc[cch] = WMMA_F16(ap1.v, bv1.v, acc[cch]);
        }

        __syncthreads();   // all reads of buffer `cur` done before it is refilled
    }

    // Normalize and store attn output (b, l, h*64+d) as f16 for the out GEMM.
#pragma unroll
    for (int r = 0; r < 8; ++r) {
        const float inv = 1.0f / lrow[r];
        const int row = q0 + r + 8 * g;
#pragma unroll
        for (int cch = 0; cch < 4; ++cch) {
            const float o = acc[cch][r] * inv;
            attn[(size_t)(b * 4096 + row) * 512 + h * 64 + cch * 16 + ln] = (_Float16)o;
        }
    }
}

// ---------------------------------------------------------------------------
// Kernel 4: output projection.  attn (8192 x 512 f16) @ Wu^T (64 x 512 f16,
// N x K) + bu -> out f32 (8192 x 64).  One wave computes a full 16x64 slab;
// K=512 -> 16 k-steps x 4 N-tiles = 64 WMMAs, A reused 4x per load.
// ---------------------------------------------------------------------------
__global__ __launch_bounds__(32)
void out_gemm(const _Float16* __restrict__ attn,
              const _Float16* __restrict__ wuT,
              const float* __restrict__ bu,
              float* __restrict__ out) {
    const int L  = threadIdx.x;
    const int g  = L >> 4;
    const int ln = L & 15;
    const int m0 = blockIdx.x * 16;

    v8f c[4] = {};
    for (int ks = 0; ks < 16; ++ks) {
        const int k0 = ks * 32;
        F16x16 a;
        const _Float16* ap = attn + (size_t)(m0 + ln) * 512 + k0 + g * 8;
        a.h2[0] = *(const v8h*)(ap);
        a.h2[1] = *(const v8h*)(ap + 16);
#pragma unroll
        for (int t = 0; t < 4; ++t) {
            F16x16 bf;
            const _Float16* bp = wuT + (size_t)(t * 16 + ln) * 512 + k0 + g * 8;
            bf.h2[0] = *(const v8h*)(bp);
            bf.h2[1] = *(const v8h*)(bp + 16);
            c[t] = WMMA_F16(a.v, bf.v, c[t]);
        }
    }

#pragma unroll
    for (int t = 0; t < 4; ++t) {
        const float bias = bu[t * 16 + ln];
#pragma unroll
        for (int r = 0; r < 8; ++r) {
            const int m = m0 + r + 8 * g;
            out[(size_t)m * 64 + t * 16 + ln] = c[t][r] + bias;
        }
    }
}

// ---------------------------------------------------------------------------
// Host launch
// ---------------------------------------------------------------------------
extern "C" void kernel_launch(void* const* d_in, const int* in_sizes, int n_in,
                              void* d_out, int out_size, void* d_ws, size_t ws_size,
                              hipStream_t stream) {
    (void)in_sizes; (void)n_in; (void)out_size; (void)ws_size;

    const float* x  = (const float*)d_in[0];  // (2,4096,64)
    const float* Wq = (const float*)d_in[1];  // (64,512)
    const float* Wk = (const float*)d_in[2];
    const float* Wv = (const float*)d_in[3];
    const float* Wu = (const float*)d_in[4];  // (512,64)
    const float* bu = (const float*)d_in[5];  // (64,)

    char* ws = (char*)d_ws;
    _Float16* xh   = (_Float16*)(ws);                       // 1 MB
    _Float16* wqT  = (_Float16*)(ws + (1u << 20));          // 64 KB each
    _Float16* wkT  = wqT + 512 * 64;
    _Float16* wvT  = wkT + 512 * 64;
    _Float16* wuT  = wvT + 512 * 64;
    _Float16* Qh   = (_Float16*)(ws + (2u << 20));          // 8 MB each
    _Float16* Kh   = Qh + (size_t)2 * 8 * 4096 * 64;
    _Float16* Vt   = Kh + (size_t)2 * 8 * 4096 * 64;
    _Float16* attn = Vt + (size_t)2 * 8 * 4096 * 64;

    cvt_f32_to_f16<<<(2 * 4096 * 64 + 255) / 256, 256, 0, stream>>>(x, xh, 2 * 4096 * 64);
    transpose_to_f16<<<(64 * 512 + 255) / 256, 256, 0, stream>>>(Wq, wqT, 64, 512);
    transpose_to_f16<<<(64 * 512 + 255) / 256, 256, 0, stream>>>(Wk, wkT, 64, 512);
    transpose_to_f16<<<(64 * 512 + 255) / 256, 256, 0, stream>>>(Wv, wvT, 64, 512);
    transpose_to_f16<<<(512 * 64 + 255) / 256, 256, 0, stream>>>(Wu, wuT, 512, 64);

    dim3 qkvGrid(512, 8);
    qkv_gemm<0><<<qkvGrid, 32, 0, stream>>>(xh, wqT, Qh);
    qkv_gemm<1><<<qkvGrid, 32, 0, stream>>>(xh, wkT, Kh);
    qkv_gemm<2><<<qkvGrid, 32, 0, stream>>>(xh, wvT, Vt);

    flash_attn<<<dim3(64, 16), 128, 0, stream>>>(Qh, Kh, Vt, attn);

    out_gemm<<<512, 32, 0, stream>>>(attn, wuT, bu, (float*)d_out);
}
